// AdaptiveContourConv_17446157156521
// MI455X (gfx1250) — matrix operand order repaired and checked by reading, not compiled, not run on hardware
//
#include <hip/hip_runtime.h>
#include <hip/hip_bf16.h>
#include <cstdint>
#include <cstddef>

// ---------------- problem constants (match reference) ----------------
#define B_    4
#define CIN   128
#define COUT  256
#define HH    64
#define WW    64
#define MID   32
#define M_TOT (B_ * HH * WW)      // 16384 output pixels
#define EPSV  1e-5f

typedef __attribute__((ext_vector_type(16))) _Float16 v16h;
typedef __attribute__((ext_vector_type(8)))  float    v8f;

union HVec { uint4 q[2]; v16h h; };

__device__ __forceinline__ float sigmoidf_(float v) { return 1.0f / (1.0f + expf(-v)); }
__device__ __forceinline__ int   iclamp(int v, int lo, int hi) { return v < lo ? lo : (v > hi ? hi : v); }

#define WMMA_F16(a, b, c) \
    __builtin_amdgcn_wmma_f32_16x16x32_f16(false, (a), false, (b), (short)0, (c), false, false)

// =====================================================================
// WMMA GEMM, 32x32 macro-tile per wave (2x2 grid of 16x16 WMMA tiles).
//   C[M x N] = A[M x K] (f16 row-major)  *  Bt[Npad x K]^T (f16, rows >= N
//   are zero padding so no masking is needed in the K loop).
// Epilogue modes:
//   0: dstF[m*ld + n] = acc + bias[n]
//   1: dstF[m*ld + n] = sigmoid(acc + bias[n])
//   2: dstH[m*ld + n + colOff] = relu(bn(acc + bias[n]))        (f16 out)
//   3: dstH[m*ld + n + colOff] = acc + bias[n]                  (f16 out)
//   4: dstF NCHW scatter: dstF[((b*ld + n)<<12) + pix] = relu(bn(acc+bias))
// =====================================================================
__global__ __launch_bounds__(32)
void wmma_gemm_kernel(const _Float16* __restrict__ A, const _Float16* __restrict__ Bt,
                      int N, int K,
                      const float* __restrict__ bias,
                      const float* __restrict__ bn_g, const float* __restrict__ bn_b,
                      const float* __restrict__ bn_m, const float* __restrict__ bn_v,
                      int mode, _Float16* __restrict__ dstH, float* __restrict__ dstF,
                      int ldDst, int colOff)
{
    const int lane  = threadIdx.x;
    const int hi    = lane >> 4;      // 0 or 1
    const int l15   = lane & 15;
    const int mBase = blockIdx.y << 5;
    const int nBase = blockIdx.x << 5;

    // A frag (16x32 f16): lanes 0-15: M=lane, K {0..7} in v0..3, {16..23} in v4..7;
    //                     lanes 16-31: same M, K shifted by +8.
    const _Float16* __restrict__ Arow0 = A + (size_t)(mBase + l15) * K + hi * 8;
    const _Float16* __restrict__ Arow1 = Arow0 + (size_t)16 * K;
    // B frag (32x16 f16): lane -> column N = lane&15; same K order as A (lanes 0-15 K {0..7,16..23}, 16-31 K +8).
    const _Float16* __restrict__ Brow0 = Bt + (size_t)(nBase + l15) * K + hi * 8;
    const _Float16* __restrict__ Brow1 = Brow0 + (size_t)16 * K;

    v8f acc[2][2] = {};
    for (int k0 = 0; k0 < K; k0 += 32) {
        HVec a0, a1, b0, b1;
        a0.q[0] = *(const uint4*)(Arow0 + k0);
        a0.q[1] = *(const uint4*)(Arow0 + k0 + 16);
        a1.q[0] = *(const uint4*)(Arow1 + k0);
        a1.q[1] = *(const uint4*)(Arow1 + k0 + 16);
        b0.q[0] = *(const uint4*)(Brow0 + k0);
        b0.q[1] = *(const uint4*)(Brow0 + k0 + 16);
        b1.q[0] = *(const uint4*)(Brow1 + k0);
        b1.q[1] = *(const uint4*)(Brow1 + k0 + 16);
        __builtin_prefetch(Arow0 + k0 + 32, 0, 1);
        __builtin_prefetch(Arow1 + k0 + 32, 0, 1);
        acc[0][0] = WMMA_F16(a0.h, b0.h, acc[0][0]);
        acc[0][1] = WMMA_F16(a0.h, b1.h, acc[0][1]);
        acc[1][0] = WMMA_F16(a1.h, b0.h, acc[1][0]);
        acc[1][1] = WMMA_F16(a1.h, b1.h, acc[1][1]);
    }

    // C/D layout: VGPR r -> row M = r + (lane>=16 ? 8 : 0), col N = lane&15
    #pragma unroll
    for (int j = 0; j < 2; ++j) {
        const int n = nBase + (j << 4) + l15;
        if (n >= N) continue;                      // zero-padded B columns: skip store
        const float bval = bias ? bias[n] : 0.0f;
        float scl = 1.0f, sub = 0.0f, beta = 0.0f;
        if (mode == 2 || mode == 4) {
            scl  = bn_g[n] * rsqrtf(bn_v[n] + EPSV);
            sub  = bn_m[n];
            beta = bn_b[n];
        }
        #pragma unroll
        for (int i = 0; i < 2; ++i) {
            #pragma unroll
            for (int r = 0; r < 8; ++r) {
                const int m = mBase + (i << 4) + r + (hi << 3);
                float val = acc[i][j][r] + bval;
                if (mode == 2 || mode == 4) {
                    val = fmaxf((val - sub) * scl + beta, 0.0f);
                } else if (mode == 1) {
                    val = sigmoidf_(val);
                }
                if (mode == 0 || mode == 1) {
                    dstF[(size_t)m * ldDst + n] = val;
                } else if (mode == 4) {
                    const int bb = m >> 12, pix = m & 4095;
                    dstF[(((size_t)(bb * ldDst + n)) << 12) + pix] = val;
                } else {
                    dstH[(size_t)m * ldDst + n + colOff] = (_Float16)val;
                }
            }
        }
    }
}

// ---------------- f32 -> f16 weight conversion with zero padding ----------------
// dst has 'total' elements; elements >= n are zero (pad rows of Bt).
__global__ void cvt_pad_f16_kernel(const float* __restrict__ src, _Float16* __restrict__ dst,
                                   int n, int total) {
    int i = blockIdx.x * blockDim.x + threadIdx.x;
    if (i < total) dst[i] = (i < n) ? (_Float16)src[i] : (_Float16)0.0f;
}

// ---------------- im2col (3x3, pad 1) from NCHW f32 ----------------
__global__ void im2col_nchw_kernel(const float* __restrict__ x, _Float16* __restrict__ A, int C) {
    int idx = blockIdx.x * blockDim.x + threadIdx.x;
    if (idx >= M_TOT * C) return;
    const int c = idx % C, m = idx / C;
    const int b = m >> 12, pix = m & 4095, y = pix >> 6, xx = pix & 63;
    const float* xp = x + (((size_t)b * C + c) << 12);
    _Float16* ap = A + (size_t)m * (C * 9) + c * 9;
    #pragma unroll
    for (int ky = 0; ky < 3; ++ky)
        #pragma unroll
        for (int kx = 0; kx < 3; ++kx) {
            const int yy = y + ky - 1, xc = xx + kx - 1;
            float v = (yy >= 0 && yy < HH && xc >= 0 && xc < WW) ? xp[(yy << 6) + xc] : 0.0f;
            ap[ky * 3 + kx] = (_Float16)v;
        }
}

// ---------------- im2col (3x3, pad 1) from NHWC f16 ----------------
__global__ void im2col_nhwc_kernel(const _Float16* __restrict__ src, _Float16* __restrict__ A, int C) {
    int idx = blockIdx.x * blockDim.x + threadIdx.x;
    if (idx >= M_TOT * C) return;
    const int c = idx % C, m = idx / C;
    const int b = m >> 12, pix = m & 4095, y = pix >> 6, xx = pix & 63;
    _Float16* ap = A + (size_t)m * (C * 9) + c * 9;
    #pragma unroll
    for (int ky = 0; ky < 3; ++ky)
        #pragma unroll
        for (int kx = 0; kx < 3; ++kx) {
            const int yy = y + ky - 1, xc = xx + kx - 1;
            _Float16 v = (_Float16)0.0f;
            if (yy >= 0 && yy < HH && xc >= 0 && xc < WW)
                v = src[(size_t)((b << 12) + (yy << 6) + xc) * C + c];
            ap[ky * 3 + kx] = v;
        }
}

// ---------------- modulated deformable sampling -> im2col A (f16) ----------------
__global__ void deform_sample_kernel(const float* __restrict__ x, const float* __restrict__ off,
                                     const float* __restrict__ msk, _Float16* __restrict__ A) {
    int idx = blockIdx.x * blockDim.x + threadIdx.x;
    if (idx >= M_TOT * 9) return;
    const int kk = idx % 9, m = idx / 9;
    const int b = m >> 12, pix = m & 4095, y = pix >> 6, xx = pix & 63;
    const float dy = off[(size_t)m * 18 + kk * 2 + 0];
    const float dx = off[(size_t)m * 18 + kk * 2 + 1];
    const float mk = msk[(size_t)m * 9 + kk];
    const float py = (float)(y - 1 + kk / 3) + dy;
    const float px = (float)(xx - 1 + kk % 3) + dx;
    const float y0f = floorf(py), x0f = floorf(px);
    const float wy1 = py - y0f,   wx1 = px - x0f;
    const float wy0 = 1.0f - wy1, wx0 = 1.0f - wx1;
    const bool vy0 = (y0f >= 0.0f)        && (y0f <= (float)(HH - 1));
    const bool vy1 = (y0f + 1.0f >= 0.0f) && (y0f + 1.0f <= (float)(HH - 1));
    const bool vx0 = (x0f >= 0.0f)        && (x0f <= (float)(WW - 1));
    const bool vx1 = (x0f + 1.0f >= 0.0f) && (x0f + 1.0f <= (float)(WW - 1));
    const int y0 = iclamp((int)y0f, 0, HH - 1),     y1 = iclamp((int)y0f + 1, 0, HH - 1);
    const int x0 = iclamp((int)x0f, 0, WW - 1),     x1 = iclamp((int)x0f + 1, 0, WW - 1);
    const float w00 = wy0 * wx0 * ((vy0 && vx0) ? mk : 0.0f);
    const float w01 = wy0 * wx1 * ((vy0 && vx1) ? mk : 0.0f);
    const float w10 = wy1 * wx0 * ((vy1 && vx0) ? mk : 0.0f);
    const float w11 = wy1 * wx1 * ((vy1 && vx1) ? mk : 0.0f);
    const int i00 = (y0 << 6) + x0, i01 = (y0 << 6) + x1;
    const int i10 = (y1 << 6) + x0, i11 = (y1 << 6) + x1;
    const float* xb = x + (((size_t)b * CIN) << 12);
    _Float16* ap = A + (size_t)m * (CIN * 9) + kk;
    for (int c = 0; c < CIN; ++c) {
        const float* xc = xb + ((size_t)c << 12);
        const float v = w00 * xc[i00] + w01 * xc[i01] + w10 * xc[i10] + w11 * xc[i11];
        ap[c * 9] = (_Float16)v;
    }
}

// ---------------- depthwise 3x3 + BN + ReLU  (NCHW f32 -> NHWC f16) ----------------
__global__ void depthwise_bn_relu_kernel(const float* __restrict__ x,
                                         const float* __restrict__ w, const float* __restrict__ bias,
                                         const float* __restrict__ g, const float* __restrict__ bb,
                                         const float* __restrict__ mean, const float* __restrict__ var,
                                         _Float16* __restrict__ hc) {
    int idx = blockIdx.x * blockDim.x + threadIdx.x;
    if (idx >= M_TOT * CIN) return;
    const int c = idx % CIN, m = idx / CIN;
    const int b = m >> 12, pix = m & 4095, y = pix >> 6, xx = pix & 63;
    const float* xp = x + (((size_t)b * CIN + c) << 12);
    const float* wp = w + c * 9;
    float acc = bias[c];
    #pragma unroll
    for (int ky = 0; ky < 3; ++ky)
        #pragma unroll
        for (int kx = 0; kx < 3; ++kx) {
            const int yy = y + ky - 1, xc = xx + kx - 1;
            if (yy >= 0 && yy < HH && xc >= 0 && xc < WW)
                acc += xp[(yy << 6) + xc] * wp[ky * 3 + kx];
        }
    const float scl = g[c] * rsqrtf(var[c] + EPSV);
    acc = fmaxf((acc - mean[c]) * scl + bb[c], 0.0f);
    hc[(size_t)m * CIN + c] = (_Float16)acc;
}

// ---------------- 7x7 spatial attention (384ch -> 1), sigmoid ----------------
__global__ void attn_kernel(const _Float16* __restrict__ comb, const float* __restrict__ sw,
                            const float* __restrict__ sb, float* __restrict__ attn) {
    int m = blockIdx.x * blockDim.x + threadIdx.x;
    if (m >= M_TOT) return;
    const int b = m >> 12, pix = m & 4095, y = pix >> 6, xx = pix & 63;
    float acc = sb[0];
    for (int ky = 0; ky < 7; ++ky) {
        const int yy = y + ky - 3;
        if (yy < 0 || yy >= HH) continue;
        for (int kx = 0; kx < 7; ++kx) {
            const int xc = xx + kx - 3;
            if (xc < 0 || xc >= WW) continue;
            const _Float16* row = comb + (size_t)((b << 12) + (yy << 6) + xc) * 384;
            const float* wp = sw + ky * 7 + kx;       // sa_w[ch][ky][kx] = sw[ch*49 + ky*7 + kx]
            for (int ch = 0; ch < 384; ++ch)
                acc += (float)row[ch] * wp[ch * 49];
        }
    }
    attn[m] = sigmoidf_(acc);
}

// ---------------- scale combined by per-pixel attention ----------------
__global__ void scale_comb_kernel(_Float16* __restrict__ comb, const float* __restrict__ attn) {
    int idx = blockIdx.x * blockDim.x + threadIdx.x;
    if (idx >= M_TOT * 384) return;
    comb[idx] = (_Float16)((float)comb[idx] * attn[idx / 384]);
}

// =====================================================================
extern "C" void kernel_launch(void* const* d_in, const int* in_sizes, int n_in,
                              void* d_out, int out_size, void* d_ws, size_t ws_size,
                              hipStream_t stream) {
    (void)in_sizes; (void)n_in; (void)out_size; (void)ws_size;
    const float* x      = (const float*)d_in[0];
    const float* oc1_w  = (const float*)d_in[1];
    const float* oc1_b  = (const float*)d_in[2];
    const float* obn_g  = (const float*)d_in[3];
    const float* obn_b  = (const float*)d_in[4];
    const float* obn_m  = (const float*)d_in[5];
    const float* obn_v  = (const float*)d_in[6];
    const float* oc2_w  = (const float*)d_in[7];
    const float* oc2_b  = (const float*)d_in[8];
    const float* mc1_w  = (const float*)d_in[9];
    const float* mc1_b  = (const float*)d_in[10];
    const float* mbn_g  = (const float*)d_in[11];
    const float* mbn_b  = (const float*)d_in[12];
    const float* mbn_m  = (const float*)d_in[13];
    const float* mbn_v  = (const float*)d_in[14];
    const float* mc2_w  = (const float*)d_in[15];
    const float* mc2_b  = (const float*)d_in[16];
    const float* dc_w   = (const float*)d_in[17];
    const float* dc_b   = (const float*)d_in[18];
    const float* cbdw_w = (const float*)d_in[19];
    const float* cbdw_b = (const float*)d_in[20];
    const float* cbn_g  = (const float*)d_in[21];
    const float* cbn_b  = (const float*)d_in[22];
    const float* cbn_m  = (const float*)d_in[23];
    const float* cbn_v  = (const float*)d_in[24];
    const float* cbpw_w = (const float*)d_in[25];
    const float* cbpw_b = (const float*)d_in[26];
    const float* sa_w   = (const float*)d_in[27];
    const float* sa_b   = (const float*)d_in[28];
    const float* fu_w   = (const float*)d_in[29];
    const float* fu_b   = (const float*)d_in[30];
    const float* fbn_g  = (const float*)d_in[31];
    const float* fbn_b  = (const float*)d_in[32];
    const float* fbn_m  = (const float*)d_in[33];
    const float* fbn_v  = (const float*)d_in[34];
    float* out = (float*)d_out;

    // ---- workspace carve (all buffers fully rewritten every call) ----
    char* p = (char*)d_ws;
    auto carve = [&](size_t bytes) -> char* {
        char* r = p;
        p += (bytes + 255) & ~(size_t)255;
        return r;
    };
    _Float16* wA_big  = (_Float16*)carve((size_t)M_TOT * 1152 * 2); // im2col(x); later reused for deform A
    _Float16* wA_mid  = (_Float16*)carve((size_t)M_TOT * 288 * 2);  // im2col(h) then im2col(hm)
    _Float16* h_mat   = (_Float16*)carve((size_t)M_TOT * MID * 2);
    _Float16* hm_mat  = (_Float16*)carve((size_t)M_TOT * MID * 2);
    float*    off_mat = (float*)   carve((size_t)M_TOT * 18 * 4);
    float*    msk_mat = (float*)   carve((size_t)M_TOT * 9 * 4);
    _Float16* hc_mat  = (_Float16*)carve((size_t)M_TOT * CIN * 2);
    _Float16* comb    = (_Float16*)carve((size_t)M_TOT * 384 * 2);
    float*    attn    = (float*)   carve((size_t)M_TOT * 4);
    // f16 weights, Bt padded to 32 rows so the GEMM K-loop never masks B
    _Float16* w_oc1   = (_Float16*)carve((size_t)MID * 1152 * 2);   // 32  (already mult of 32)
    _Float16* w_oc2   = (_Float16*)carve((size_t)32 * 288 * 2);     // 18 -> 32
    _Float16* w_mc1   = (_Float16*)carve((size_t)MID * 1152 * 2);
    _Float16* w_mc2   = (_Float16*)carve((size_t)32 * 288 * 2);     // 9 -> 32
    _Float16* w_dc    = (_Float16*)carve((size_t)COUT * 1152 * 2);
    _Float16* w_pw    = (_Float16*)carve((size_t)CIN * CIN * 2);
    _Float16* w_fu    = (_Float16*)carve((size_t)COUT * 384 * 2);

    const int TB = 256;
    auto blocks = [](int n, int tb) { return (n + tb - 1) / tb; };
    auto cvt = [&](const float* s, _Float16* d, int n, int total) {
        cvt_pad_f16_kernel<<<blocks(total, TB), TB, 0, stream>>>(s, d, n, total);
    };
    auto gemm = [&](const _Float16* A, const _Float16* Bt, int N, int K, const float* bias,
                    const float* g, const float* bb, const float* mn, const float* vr,
                    int mode, _Float16* dH, float* dF, int ld, int co) {
        dim3 grid((N + 31) / 32, M_TOT / 32);
        wmma_gemm_kernel<<<grid, dim3(32), 0, stream>>>(A, Bt, N, K, bias, g, bb, mn, vr,
                                                        mode, dH, dF, ld, co);
    };

    // 1) weights -> f16 (natural OIHW layout == Bt[N][K] with K = c*9+kk), zero-padded rows
    cvt(oc1_w, w_oc1, MID * 1152, MID * 1152);
    cvt(oc2_w, w_oc2, 18 * 288, 32 * 288);
    cvt(mc1_w, w_mc1, MID * 1152, MID * 1152);
    cvt(mc2_w, w_mc2, 9 * 288, 32 * 288);
    cvt(dc_w, w_dc, COUT * 1152, COUT * 1152);
    cvt(cbpw_w, w_pw, CIN * CIN, CIN * CIN);
    cvt(fu_w, w_fu, COUT * 384, COUT * 384);

    // 2) im2col(x) once; shared by offset and mask first convs
    im2col_nchw_kernel<<<blocks(M_TOT * CIN, TB), TB, 0, stream>>>(x, wA_big, CIN);

    // 3) h = relu(bn(conv(x, oc1)))  -> NHWC f16
    gemm(wA_big, w_oc1, MID, 1152, oc1_b, obn_g, obn_b, obn_m, obn_v, 2, h_mat, nullptr, MID, 0);
    // 4) hm = relu(bn(conv(x, mc1))) -> NHWC f16
    gemm(wA_big, w_mc1, MID, 1152, mc1_b, mbn_g, mbn_b, mbn_m, mbn_v, 2, hm_mat, nullptr, MID, 0);

    // 5) offset = conv(h, oc2) + b   -> f32 [M][18]
    im2col_nhwc_kernel<<<blocks(M_TOT * MID, TB), TB, 0, stream>>>(h_mat, wA_mid, MID);
    gemm(wA_mid, w_oc2, 18, 288, oc2_b, nullptr, nullptr, nullptr, nullptr, 0, nullptr, off_mat, 18, 0);

    // 6) mask = sigmoid(conv(hm, mc2) + b) -> f32 [M][9]
    im2col_nhwc_kernel<<<blocks(M_TOT * MID, TB), TB, 0, stream>>>(hm_mat, wA_mid, MID);
    gemm(wA_mid, w_mc2, 9, 288, mc2_b, nullptr, nullptr, nullptr, nullptr, 1, nullptr, msk_mat, 9, 0);

    // 7) bilinear-gather * mask -> deform im2col A (reuses wA_big)
    deform_sample_kernel<<<blocks(M_TOT * 9, TB), TB, 0, stream>>>(x, off_mat, msk_mat, wA_big);

    // 8) main_feat = A_def x dc_w + dc_b -> comb[:, 0:256] f16
    gemm(wA_big, w_dc, COUT, 1152, dc_b, nullptr, nullptr, nullptr, nullptr, 3, comb, nullptr, 384, 0);

    // 9) contour: depthwise+BN+ReLU then pointwise -> comb[:, 256:384]
    depthwise_bn_relu_kernel<<<blocks(M_TOT * CIN, TB), TB, 0, stream>>>(
        x, cbdw_w, cbdw_b, cbn_g, cbn_b, cbn_m, cbn_v, hc_mat);
    gemm(hc_mat, w_pw, CIN, CIN, cbpw_b, nullptr, nullptr, nullptr, nullptr, 3, comb, nullptr, 384, 256);

    // 10) spatial attention (7x7, pad 3) + sigmoid, then scale combined
    attn_kernel<<<blocks(M_TOT, TB), TB, 0, stream>>>(comb, sa_w, sa_b, attn);
    scale_comb_kernel<<<blocks(M_TOT * 384, TB), TB, 0, stream>>>(comb, attn);

    // 11) fusion 1x1 + BN + ReLU -> d_out (NCHW f32)
    gemm(comb, w_fu, COUT, 384, fu_b, fbn_g, fbn_b, fbn_m, fbn_v, 4, nullptr, out, COUT, 0);
}
